// JetClassifier_57234734186744
// MI455X (gfx1250) — matrix operand order repaired and checked by reading, not compiled, not run on hardware
//
#include <hip/hip_runtime.h>

typedef __attribute__((ext_vector_type(16))) _Float16 v16h;
typedef __attribute__((ext_vector_type(8)))  float    v8f;

#define G_    2048
#define NPG_  32
#define H_    64
#define EMB_  5
#define FJ_   16
#define EPG_  512
#define EIN_  205
#define NIN_  134
#define KT_E  7    /* K padded to 224 */
#define KT_N  5    /* K padded to 160 */

__global__ __launch_bounds__(256)
void jet_graph_kernel(const int* __restrict__ esrc, const int* __restrict__ edst,
                      const float* __restrict__ h0, const float* __restrict__ npred,
                      const float* __restrict__ nemb, const float* __restrict__ epred,
                      const float* __restrict__ jet,
                      const float* __restrict__ Weg, const float* __restrict__ beg,
                      const float* __restrict__ Wng, const float* __restrict__ bng,
                      const float* __restrict__ Wc0, const float* __restrict__ bc0,
                      const float* __restrict__ Wc1, const float* __restrict__ bc1,
                      const float* __restrict__ Wc2, const float* __restrict__ bc2,
                      float* __restrict__ out)
{
  __shared__ float    h32S[NPG_ * H_];        // current node features (f32)
  __shared__ float    aggS[NPG_ * H_];        // mailbox sum
  __shared__ float    embS[NPG_ * 8];         // node_emb padded
  __shared__ float    clsS[NPG_];             // argmax(node_pred) as float
  __shared__ _Float16 sigepH[EPG_];           // sigmoid(edge_pred)
  __shared__ _Float16 mH[H_];                 // per-graph mean(h) in f16
  __shared__ _Float16 Psh[NPG_ * 72];         // [h(64), cls(1), emb(5)] per node, padded
  __shared__ _Float16 WeB[KT_E * 4 * 32 * 16];// We in B-fragment order
  __shared__ _Float16 WnB[KT_N * 4 * 32 * 16];// Wn in B-fragment order
  __shared__ int      slocS[EPG_];
  __shared__ int      dlocS[EPG_];
  __shared__ float    y0S[80];
  __shared__ float    y1S[128];
  __shared__ float    y2S[64];

  const int tid   = threadIdx.x;
  const int g     = blockIdx.x;
  const int wave  = tid >> 5;
  const int lane  = tid & 31;
  const int lrow  = lane & 15;
  const int lhalf = lane >> 4;

  // ---------------- initial staging ----------------
  for (int idx = tid; idx < NPG_ * H_; idx += 256) h32S[idx] = h0[g * NPG_ * H_ + idx];
  if (tid < NPG_) {
    const float* p = npred + (g * NPG_ + tid) * 4;
    int b = 0; float bv = p[0];
    #pragma unroll
    for (int c = 1; c < 4; ++c) { float v = p[c]; if (v > bv) { bv = v; b = c; } }
    clsS[tid] = (float)b;
  }
  for (int idx = tid; idx < NPG_ * EMB_; idx += 256) {
    int n = idx / EMB_, k = idx - n * EMB_;
    embS[n * 8 + k] = nemb[(g * NPG_ + n) * EMB_ + k];
  }
  for (int idx = tid; idx < EPG_; idx += 256) {
    float x = epred[g * EPG_ + idx];
    sigepH[idx] = (_Float16)(1.0f / (1.0f + expf(-x)));
    slocS[idx]  = esrc[g * EPG_ + idx] - g * NPG_;
    dlocS[idx]  = edst[g * EPG_ + idx] - g * NPG_;
  }

  // edge feature k -> value (piecewise over P / meanh / sigep)
  auto efetch = [&](int k, int s, int d, float sg) -> _Float16 {
    if (k == 0)   return (_Float16)sg;
    if (k < 66)   return Psh[s * 72 + (k - 1)];
    if (k < 136)  return Psh[d * 72 + (k - 66)];
    if (k < 141)  return Psh[s * 72 + 65 + (k - 136)];
    if (k < EIN_) return mH[k - 141];
    return (_Float16)0.0f;
  };
  // node-update feature k
  auto nfetch = [&](int k, int node) -> _Float16 {
    if (k < 64)   return (_Float16)h32S[node * H_ + k];
    if (k < 128)  return (_Float16)aggS[node * H_ + (k - 64)];
    if (k < 133)  return (_Float16)embS[node * 8 + (k - 128)];
    if (k == 133) return (_Float16)clsS[node];
    return (_Float16)0.0f;
  };

  // ---------------- 2 message-passing blocks ----------------
  for (int blk = 0; blk < 2; ++blk) {
    __syncthreads();
    // per-graph mean of h (f16 for edge features)
    if (tid < H_) {
      float s = 0.0f;
      #pragma unroll
      for (int n = 0; n < NPG_; ++n) s += h32S[n * H_ + tid];
      mH[tid] = (_Float16)(s * (1.0f / 32.0f));
    }
    // pack per-node feature rows P = [h, cls, emb] in f16
    for (int idx = tid; idx < NPG_ * 70; idx += 256) {
      int n = idx / 70, k = idx - n * 70;
      _Float16 v;
      if (k < 64)       v = (_Float16)h32S[n * H_ + k];
      else if (k == 64) v = (_Float16)clsS[n];
      else              v = (_Float16)embS[n * 8 + (k - 65)];
      Psh[n * 72 + k] = v;
    }
    for (int idx = tid; idx < NPG_ * H_; idx += 256) aggS[idx] = 0.0f;
    // swizzle We[blk] into B-fragment layout (lane-contiguous 16 halves)
    for (int idx = tid; idx < KT_E * 4 * 32 * 16; idx += 256) {
      int j = idx & 15, l = (idx >> 4) & 31, n = (idx >> 9) & 3, t = idx >> 11;
      int K = t * 32 + ((l >= 16) ? 8 : 0) + (j & 7) + ((j & 8) ? 16 : 0);
      int col = n * 16 + (l & 15);
      WeB[idx] = (K < EIN_) ? (_Float16)Weg[(blk * EIN_ + K) * H_ + col] : (_Float16)0.0f;
    }
    for (int idx = tid; idx < KT_N * 4 * 32 * 16; idx += 256) {
      int j = idx & 15, l = (idx >> 4) & 31, n = (idx >> 9) & 3, t = idx >> 11;
      int K = t * 32 + ((l >= 16) ? 8 : 0) + (j & 7) + ((j & 8) ? 16 : 0);
      int col = n * 16 + (l & 15);
      WnB[idx] = (K < NIN_) ? (_Float16)Wng[(blk * NIN_ + K) * H_ + col] : (_Float16)0.0f;
    }
    __syncthreads();

    // ---- edge phase: each wave handles 64 edges (4 M-tiles of 16) ----
    float beL[4];
    #pragma unroll
    for (int n = 0; n < 4; ++n) beL[n] = beg[blk * H_ + n * 16 + lrow];

    for (int mt = 0; mt < 4; ++mt) {
      const int ebase = wave * 64 + mt * 16;
      const int erow  = ebase + lrow;
      const int s     = slocS[erow];
      const int d     = dlocS[erow];
      const float sg  = (float)sigepH[erow];
      v8f acc0 = {}, acc1 = {}, acc2 = {}, acc3 = {};
      #pragma unroll
      for (int t = 0; t < KT_E; ++t) {
        v16h a;
        const int kb = t * 32 + lhalf * 8;   // A-fragment K layout (16-bit A 16x32)
        #pragma unroll
        for (int j = 0; j < 8; ++j) {
          a[j]     = efetch(kb + j,      s, d, sg);
          a[j + 8] = efetch(kb + 16 + j, s, d, sg);
        }
        const v16h b0 = *(const v16h*)&WeB[((t * 4 + 0) * 32 + lane) * 16];
        const v16h b1 = *(const v16h*)&WeB[((t * 4 + 1) * 32 + lane) * 16];
        const v16h b2 = *(const v16h*)&WeB[((t * 4 + 2) * 32 + lane) * 16];
        const v16h b3 = *(const v16h*)&WeB[((t * 4 + 3) * 32 + lane) * 16];
        acc0 = __builtin_amdgcn_wmma_f32_16x16x32_f16(false, a, false, b0, (short)0, acc0, false, false);
        acc1 = __builtin_amdgcn_wmma_f32_16x16x32_f16(false, a, false, b1, (short)0, acc1, false, false);
        acc2 = __builtin_amdgcn_wmma_f32_16x16x32_f16(false, a, false, b2, (short)0, acc2, false, false);
        acc3 = __builtin_amdgcn_wmma_f32_16x16x32_f16(false, a, false, b3, (short)0, acc3, false, false);
      }
      // tanh + scatter into LDS mailbox (C/D layout: VGPR r -> M = r + 8*lhalf, N = lane%16)
      #pragma unroll
      for (int r = 0; r < 8; ++r) {
        const int er = ebase + r + 8 * lhalf;
        const int dd = dlocS[er];
        float* aggRow = &aggS[dd * H_ + lrow];
        atomicAdd(aggRow +  0, tanhf(acc0[r] + beL[0]));
        atomicAdd(aggRow + 16, tanhf(acc1[r] + beL[1]));
        atomicAdd(aggRow + 32, tanhf(acc2[r] + beL[2]));
        atomicAdd(aggRow + 48, tanhf(acc3[r] + beL[3]));
      }
    }
    __syncthreads();

    // ---- node phase: h = relu([h, agg, emb, cls] @ Wn + bn); waves 0,1 ----
    if (wave < 2) {
      const int nbase = wave * 16;
      const int node  = nbase + lrow;
      float bnL[4];
      #pragma unroll
      for (int n = 0; n < 4; ++n) bnL[n] = bng[blk * H_ + n * 16 + lrow];
      v8f acc0 = {}, acc1 = {}, acc2 = {}, acc3 = {};
      #pragma unroll
      for (int t = 0; t < KT_N; ++t) {
        v16h a;
        const int kb = t * 32 + lhalf * 8;
        #pragma unroll
        for (int j = 0; j < 8; ++j) {
          a[j]     = nfetch(kb + j,      node);
          a[j + 8] = nfetch(kb + 16 + j, node);
        }
        const v16h b0 = *(const v16h*)&WnB[((t * 4 + 0) * 32 + lane) * 16];
        const v16h b1 = *(const v16h*)&WnB[((t * 4 + 1) * 32 + lane) * 16];
        const v16h b2 = *(const v16h*)&WnB[((t * 4 + 2) * 32 + lane) * 16];
        const v16h b3 = *(const v16h*)&WnB[((t * 4 + 3) * 32 + lane) * 16];
        acc0 = __builtin_amdgcn_wmma_f32_16x16x32_f16(false, a, false, b0, (short)0, acc0, false, false);
        acc1 = __builtin_amdgcn_wmma_f32_16x16x32_f16(false, a, false, b1, (short)0, acc1, false, false);
        acc2 = __builtin_amdgcn_wmma_f32_16x16x32_f16(false, a, false, b2, (short)0, acc2, false, false);
        acc3 = __builtin_amdgcn_wmma_f32_16x16x32_f16(false, a, false, b3, (short)0, acc3, false, false);
      }
      #pragma unroll
      for (int r = 0; r < 8; ++r) {
        const int nd = nbase + r + 8 * lhalf;
        h32S[nd * H_ + lrow +  0] = fmaxf(acc0[r] + bnL[0], 0.0f);
        h32S[nd * H_ + lrow + 16] = fmaxf(acc1[r] + bnL[1], 0.0f);
        h32S[nd * H_ + lrow + 32] = fmaxf(acc2[r] + bnL[2], 0.0f);
        h32S[nd * H_ + lrow + 48] = fmaxf(acc3[r] + bnL[3], 0.0f);
      }
    }
  }

  // ---------------- classifier head ----------------
  __syncthreads();
  if (tid < H_) {
    float s = 0.0f;
    #pragma unroll
    for (int n = 0; n < NPG_; ++n) s += h32S[n * H_ + tid];
    y0S[tid] = s * (1.0f / 32.0f);
  }
  if (tid >= H_ && tid < H_ + FJ_) y0S[tid] = jet[g * FJ_ + (tid - H_)];
  __syncthreads();
  if (tid < 128) {                       // no activation after first layer
    float s = bc0[tid];
    for (int k = 0; k < 80; ++k) s += y0S[k] * Wc0[k * 128 + tid];
    y1S[tid] = s;
  }
  __syncthreads();
  if (tid < 64) {
    float s = bc1[tid];
    for (int k = 0; k < 128; ++k) s += y1S[k] * Wc1[k * 64 + tid];
    y2S[tid] = fmaxf(s, 0.0f);
  }
  __syncthreads();
  if (tid < 3) {
    float s = bc2[tid];
    for (int k = 0; k < 64; ++k) s += y2S[k] * Wc2[k * 3 + tid];
    out[g * 3 + tid] = s;
  }
}

extern "C" void kernel_launch(void* const* d_in, const int* in_sizes, int n_in,
                              void* d_out, int out_size, void* d_ws, size_t ws_size,
                              hipStream_t stream) {
  (void)in_sizes; (void)n_in; (void)d_ws; (void)ws_size; (void)out_size;
  const int*   esrc  = (const int*)  d_in[0];
  const int*   edst  = (const int*)  d_in[1];
  /* d_in[2] graph_id implicit in layout */
  const float* h0    = (const float*)d_in[3];
  const float* npred = (const float*)d_in[4];
  const float* nemb  = (const float*)d_in[5];
  const float* epred = (const float*)d_in[6];
  const float* jet   = (const float*)d_in[7];
  const float* We    = (const float*)d_in[8];
  const float* be    = (const float*)d_in[9];
  const float* Wn    = (const float*)d_in[10];
  const float* bn    = (const float*)d_in[11];
  const float* Wc0   = (const float*)d_in[12];
  const float* bc0   = (const float*)d_in[13];
  const float* Wc1   = (const float*)d_in[14];
  const float* bc1   = (const float*)d_in[15];
  const float* Wc2   = (const float*)d_in[16];
  const float* bc2   = (const float*)d_in[17];

  jet_graph_kernel<<<dim3(G_), dim3(256), 0, stream>>>(
      esrc, edst, h0, npred, nemb, epred, jet,
      We, be, Wn, bn, Wc0, bc0, Wc1, bc1, Wc2, bc2, (float*)d_out);
}
